// Rvt_59605556134239
// MI455X (gfx1250) — matrix operand order repaired and checked by reading, not compiled, not run on hardware
//
#include <hip/hip_runtime.h>

// ---------------------------------------------------------------------------
// MI455X (gfx1250): all GEMM stages on v_wmma_f32_16x16x32_bf16 (wave32).
// Block: 256 threads / 8 waves; tile 128(M) x 128(N); wave tile 32x64
// (2x4 accumulators -> 8 WMMAs per K-step). Register-prefetch + LDS double
// buffering, one barrier per K-step, all 6 fragments preloaded per step so
// ds_load latency is overlapped across the 8 WMMAs.
// ---------------------------------------------------------------------------

typedef __attribute__((ext_vector_type(16))) __bf16       v16bf;
typedef __attribute__((ext_vector_type(8)))  float        v8f;
typedef __attribute__((ext_vector_type(4)))  unsigned int ux4;
typedef __attribute__((ext_vector_type(4)))  float        f32x4;

__device__ __forceinline__ unsigned short f2bfbits(float f) {
  unsigned u = __builtin_bit_cast(unsigned, f);
  u = (u + 0x7FFFu + ((u >> 16) & 1u)) >> 16;   // round-to-nearest-even
  return (unsigned short)u;
}
__device__ __forceinline__ __bf16 f2bf(float f) {
  unsigned short h = f2bfbits(f);
  return __builtin_bit_cast(__bf16, h);
}
__device__ __forceinline__ float bfbits2f(unsigned short h) {
  unsigned u = ((unsigned)h) << 16;
  return __builtin_bit_cast(float, u);
}
// Branchless tanh: 1 - 2/(exp(2x)+1).  exp->inf => 1; exp->0 => -1. No NaN.
__device__ __forceinline__ float fast_tanh(float x) {
  float e = __expf(2.0f * x);
  return 1.0f - 2.0f / (e + 1.0f);
}
__device__ __forceinline__ float fast_sigmoid(float x) {
  return 1.0f / (1.0f + __expf(-x));
}

// LDS tile strides (bf16 elements). 40*2B = 80B row stride -> 16B aligned.
#define LDSA 40
#define LDSB 40

// A fragment (16x32 bf16): lane l -> row l&15; K chunks [kb,kb+8) and
// [kb+16,kb+24), kb = (l>>4)*8.  (ISA 7.12.2, 16-bit A 16x32)
__device__ __forceinline__ v16bf load_fragA(const __bf16* base, int lane) {
  const __bf16* row = base + (lane & 15) * LDSA;
  int kb = (lane >> 4) * 8;
  union { v16bf f; ux4 q[2]; } u;
  u.q[0] = *(const ux4*)(row + kb);
  u.q[1] = *(const ux4*)(row + kb + 16);
  return u.f;
}
// B fragment from LDS stored [N][K]: lane l -> col l&15; K chunks [kb,kb+8)
// and [kb+16,kb+24), kb = (l>>4)*8.
__device__ __forceinline__ v16bf load_fragB(const __bf16* base, int lane) {
  const __bf16* row = base + (lane & 15) * LDSB;
  int kb = (lane >> 4) * 8;
  union { v16bf f; ux4 q[2]; } u;
  u.q[0] = *(const ux4*)(row + kb);
  u.q[1] = *(const ux4*)(row + kb + 16);
  return u.f;
}
__device__ __forceinline__ v8f wmma_bf16(v16bf a, v16bf b, v8f c) {
  return __builtin_amdgcn_wmma_f32_16x16x32_bf16(
      false, a, false, b, (short)0, c, false, false);
}

// ---------------------------------------------------------------------------
// D[M x N] = A[M x K] * B[N x K]^T  (B stored [N][K]).
// AMODE: 0 = A bf16 row-major (lda), batched by sA
//        1 = A fp32 NCHW gather-transpose (stage 1)
//        2 = A bf16 pixel-major with 3x3 spatial shift (implicit conv)
// BMODE: 0 = B bf16 [N][K] (ldb); 1 = B bf16 [K][N], transposed while staging
// EPI:   0 = bf16 store, 1 = tanh->bf16, 2 = f32 store,
//        3 = +bias, f32 NCHW store (ldd = HW = 1024)
// ---------------------------------------------------------------------------
template <int AMODE, int BMODE, int EPI>
__global__ __launch_bounds__(256)
void gemm_wmma(const void* __restrict__ Ap, const __bf16* __restrict__ Bp,
               void* __restrict__ Dp, int M, int N, int K,
               int lda, int ldb, int ldd,
               long long sA, long long sB, long long sD,
               const float* __restrict__ bias) {
  __shared__ __align__(16) __bf16 As[2][128 * LDSA];
  __shared__ __align__(16) __bf16 Bs[2][128 * LDSB];

  const int t    = threadIdx.x;
  const int lane = t & 31;
  const int wid  = t >> 5;
  const int wm   = wid >> 1;             // 0..3 -> 32-row M sub-tile
  const int wn   = wid & 1;              // 0..1 -> 64-col N sub-tile
  const int mtile = blockIdx.y * 128;
  const int ntile = blockIdx.x * 128;
  const long long bz = blockIdx.z;

  const __bf16* Abf = nullptr;
  const float*  Af  = nullptr;
  if constexpr (AMODE == 1) Af = (const float*)Ap;
  else Abf = (const __bf16*)Ap + bz * sA;
  const __bf16* B = Bp + bz * sB;

  // ---- prefetch helpers (registers <-> global) ----
  auto prefA = [&](int kq, ux4& r0, ux4& r1, f32x4* pf) {
    if constexpr (AMODE == 0) {
      int row = t >> 1, seg = (t & 1) * 16;
      const __bf16* src = Abf + (long long)(mtile + row) * lda + kq + seg;
      r0 = *(const ux4*)src;
      r1 = *(const ux4*)(src + 8);
    } else if constexpr (AMODE == 1) {
      int kk = t >> 3, mm = (t & 7) * 16;
      int n_img = mtile >> 10;
      int hwb   = (mtile & 1023) + mm;
      const float* src = Af + ((long long)n_img * K + (kq + kk)) * lda + hwb;
#pragma unroll
      for (int i = 0; i < 4; ++i) pf[i] = *(const f32x4*)(src + 4 * i);
    } else {  // AMODE == 2 : 3x3-shifted rows, zero-fill outside image
      int row = t >> 1, seg = (t & 1) * 16;
      int p = mtile + row;
      int n_img = p >> 10, hw = p & 1023;
      int y = hw >> 5, x = hw & 31;
      int tap = kq >> 8, ci0 = kq & 255;
      int dy = tap / 3 - 1, dx = tap % 3 - 1;
      int ys = y + dy, xs = x + dx;
      r0 = ux4{0u, 0u, 0u, 0u};
      r1 = ux4{0u, 0u, 0u, 0u};
      if ((unsigned)ys < 32u && (unsigned)xs < 32u) {
        const __bf16* src =
            Abf + ((long long)(n_img * 1024 + ys * 32 + xs)) * lda + ci0 + seg;
        r0 = *(const ux4*)src;
        r1 = *(const ux4*)(src + 8);
      }
    }
  };
  auto storeA = [&](__bf16* dst, ux4 r0, ux4 r1, const f32x4* pf) {
    if constexpr (AMODE == 1) {
      int kk = t >> 3, mm = (t & 7) * 16;
#pragma unroll
      for (int i = 0; i < 4; ++i) {
        dst[(mm + 4 * i + 0) * LDSA + kk] = f2bf(pf[i][0]);
        dst[(mm + 4 * i + 1) * LDSA + kk] = f2bf(pf[i][1]);
        dst[(mm + 4 * i + 2) * LDSA + kk] = f2bf(pf[i][2]);
        dst[(mm + 4 * i + 3) * LDSA + kk] = f2bf(pf[i][3]);
      }
    } else {
      int row = t >> 1, seg = (t & 1) * 16;
      *(ux4*)&dst[row * LDSA + seg]     = r0;
      *(ux4*)&dst[row * LDSA + seg + 8] = r1;
    }
  };
  auto prefB = [&](int kq, ux4& r0, ux4& r1) {
    if constexpr (BMODE == 0) {
      int row = t >> 1, seg = (t & 1) * 16;
      const __bf16* src = B + (long long)(ntile + row) * ldb + kq + seg;
      r0 = *(const ux4*)src;
      r1 = *(const ux4*)(src + 8);
    } else {  // [K][N] -> 16 contiguous N elems, transposed on store
      int kk = t >> 3, nn = (t & 7) * 16;
      const __bf16* src = B + (long long)(kq + kk) * ldb + ntile + nn;
      r0 = *(const ux4*)src;
      r1 = *(const ux4*)(src + 8);
    }
  };
  auto storeB = [&](__bf16* dst, ux4 r0, ux4 r1) {
    if constexpr (BMODE == 0) {
      int row = t >> 1, seg = (t & 1) * 16;
      *(ux4*)&dst[row * LDSB + seg]     = r0;
      *(ux4*)&dst[row * LDSB + seg + 8] = r1;
    } else {
      int kk = t >> 3, nn = (t & 7) * 16;
      union { ux4 q[2]; unsigned short h[16]; } u;
      u.q[0] = r0; u.q[1] = r1;
#pragma unroll
      for (int i = 0; i < 16; ++i)
        dst[(nn + i) * LDSB + kk] = __builtin_bit_cast(__bf16, u.h[i]);
    }
  };

  const v8f vzero = {0.f, 0.f, 0.f, 0.f, 0.f, 0.f, 0.f, 0.f};
  v8f acc[2][4] = {{vzero, vzero, vzero, vzero}, {vzero, vzero, vzero, vzero}};

  // prologue: stage k0 = 0 into buffer 0
  {
    ux4 a0{}, a1{}, b0{}, b1{};
    f32x4 pf[4];
    prefA(0, a0, a1, pf);
    prefB(0, b0, b1);
    storeA(As[0], a0, a1, pf);
    storeB(Bs[0], b0, b1);
  }
  __syncthreads();

  int cur = 0;
  for (int k0 = 0; k0 < K; k0 += 32) {
    const bool hn = (k0 + 32) < K;
    ux4 a0{}, a1{}, b0{}, b1{};
    f32x4 pf[4];
    if (hn) {           // issue next tile's global loads (latency hidden)
      prefA(k0 + 32, a0, a1, pf);
      prefB(k0 + 32, b0, b1);
    }
    // compute from buffer `cur`: preload ALL fragments into distinct regs so
    // the backend can issue every ds_load_b128 before the first WMMA and wait
    // incrementally instead of serializing per WMMA pair.
    const __bf16* Ac = As[cur];
    const __bf16* Bc = Bs[cur];
    v16bf fa0 = load_fragA(Ac + (wm * 32)      * LDSA, lane);
    v16bf fa1 = load_fragA(Ac + (wm * 32 + 16) * LDSA, lane);
    v16bf fb0 = load_fragB(Bc + (wn * 64)      * LDSB, lane);
    v16bf fb1 = load_fragB(Bc + (wn * 64 + 16) * LDSB, lane);
    v16bf fb2 = load_fragB(Bc + (wn * 64 + 32) * LDSB, lane);
    v16bf fb3 = load_fragB(Bc + (wn * 64 + 48) * LDSB, lane);
    acc[0][0] = wmma_bf16(fa0, fb0, acc[0][0]);
    acc[1][0] = wmma_bf16(fa1, fb0, acc[1][0]);
    acc[0][1] = wmma_bf16(fa0, fb1, acc[0][1]);
    acc[1][1] = wmma_bf16(fa1, fb1, acc[1][1]);
    acc[0][2] = wmma_bf16(fa0, fb2, acc[0][2]);
    acc[1][2] = wmma_bf16(fa1, fb2, acc[1][2]);
    acc[0][3] = wmma_bf16(fa0, fb3, acc[0][3]);
    acc[1][3] = wmma_bf16(fa1, fb3, acc[1][3]);
    if (hn) {           // write prefetched tile into the other buffer
      storeA(As[cur ^ 1], a0, a1, pf);
      storeB(Bs[cur ^ 1], b0, b1);
    }
    __syncthreads();    // single barrier per K-step
    cur ^= 1;
  }

  // ---------------- epilogue ----------------
  const int colL = lane & 15;
  const int hi   = lane >> 4;
#pragma unroll
  for (int i = 0; i < 2; ++i) {
#pragma unroll
    for (int j = 0; j < 4; ++j) {
      int fm = mtile + wm * 32 + i * 16 + hi * 8;   // + v gives row
      int fn = ntile + wn * 64 + j * 16 + colL;
#pragma unroll
      for (int v = 0; v < 8; ++v) {
        float val = acc[i][j][v];
        long long row = fm + v;
        if constexpr (EPI == 0) {
          __bf16* D = (__bf16*)Dp + bz * sD;
          D[row * ldd + fn] = f2bf(val);
        } else if constexpr (EPI == 1) {
          __bf16* D = (__bf16*)Dp + bz * sD;
          D[row * ldd + fn] = f2bf(fast_tanh(val));
        } else if constexpr (EPI == 2) {
          float* D = (float*)Dp + bz * sD;
          D[row * ldd + fn] = val;
        } else {  // EPI == 3: + bias, f32 NCHW store (ldd = HW, power of 2)
          float* D = (float*)Dp;
          long long nimg = row >> 10;
          long long hw   = row & (long long)(ldd - 1);
          D[(nimg * N + fn) * ldd + hw] = val + bias[fn];
        }
      }
    }
  }
}

// ---------------------------------------------------------------------------
// Small helper kernels
// ---------------------------------------------------------------------------
__global__ __launch_bounds__(256)
void pack_bf16(const float* __restrict__ in, __bf16* __restrict__ out, int n) {
  int i = blockIdx.x * 256 + threadIdx.x;
  if (i < n) out[i] = f2bf(in[i]);
}

// w_conv [256][512][3][3] -> packed [co][tap][ci<256] bf16 (h-channels dead)
__global__ __launch_bounds__(256)
void pack_conv9(const float* __restrict__ w, __bf16* __restrict__ out) {
  int i = blockIdx.x * 256 + threadIdx.x;
  if (i >= 256 * 9 * 256) return;
  int co = i / 2304, r = i % 2304, tap = r >> 8, ci = r & 255;
  out[i] = f2bf(w[((long long)co * 512 + ci) * 9 + tap]);
}

// Per-pixel L2-normalize over 256 channels (bf16 in place); one wave / pixel.
__global__ __launch_bounds__(256)
void l2norm_scale(__bf16* __restrict__ t, const float* __restrict__ s) {
  int wid = threadIdx.x >> 5, lane = threadIdx.x & 31;
  long long pix = (long long)blockIdx.x * 8 + wid;
  __bf16* row = t + pix * 256;
  union { ux4 q; unsigned short h[8]; } u;
  u.q = *(const ux4*)(row + lane * 8);
  float v[8], ss = 0.f;
#pragma unroll
  for (int i = 0; i < 8; ++i) { v[i] = bfbits2f(u.h[i]); ss += v[i] * v[i]; }
#pragma unroll
  for (int off = 16; off > 0; off >>= 1) ss += __shfl_xor(ss, off, 32);
  float r = rsqrtf(fmaxf(ss, 1e-24f));
#pragma unroll
  for (int i = 0; i < 8; ++i) {
    float sc = s ? s[lane * 8 + i] : 1.0f;
    u.h[i] = f2bfbits(v[i] * r * sc);
  }
  *(ux4*)(row + lane * 8) = u.q;
}

// Row softmax: 1024 f32 in -> 1024 bf16 out; one 256-thread block per row.
__global__ __launch_bounds__(256)
void softmax_row(const float* __restrict__ S, __bf16* __restrict__ P) {
  __shared__ float red[8];
  long long row = blockIdx.x;
  const float* in = S + row * 1024;
  __bf16* out = P + row * 1024;
  int t = threadIdx.x;
  f32x4 v = *(const f32x4*)(in + t * 4);
  float m = fmaxf(fmaxf(v[0], v[1]), fmaxf(v[2], v[3]));
#pragma unroll
  for (int off = 16; off > 0; off >>= 1) m = fmaxf(m, __shfl_xor(m, off, 32));
  if ((t & 31) == 0) red[t >> 5] = m;
  __syncthreads();
  float bm = red[0];
#pragma unroll
  for (int i = 1; i < 8; ++i) bm = fmaxf(bm, red[i]);
  float e[4], sum = 0.f;
#pragma unroll
  for (int i = 0; i < 4; ++i) { e[i] = __expf(v[i] - bm); sum += e[i]; }
#pragma unroll
  for (int off = 16; off > 0; off >>= 1) sum += __shfl_xor(sum, off, 32);
  __syncthreads();
  if ((t & 31) == 0) red[t >> 5] = sum;
  __syncthreads();
  float bs = 0.f;
#pragma unroll
  for (int i = 0; i < 8; ++i) bs += red[i];
  float inv = 1.f / bs;
  union { unsigned long long w; unsigned short h[4]; } o;
#pragma unroll
  for (int i = 0; i < 4; ++i) o.h[i] = f2bfbits(e[i] * inv);
  *(unsigned long long*)(out + t * 4) = o.w;
}

// h = (1 - sigmoid(zpre + bz[c])) * tanh(npre + bn[c]); 8 channels / thread.
__global__ __launch_bounds__(256)
void gru_combine(const __bf16* __restrict__ zp, const __bf16* __restrict__ np,
                 const float* __restrict__ bz, const float* __restrict__ bn,
                 __bf16* __restrict__ h) {
  long long i8 = ((long long)blockIdx.x * 256 + threadIdx.x) * 8;
  int c0 = (int)(i8 & 255);
  union { ux4 q; unsigned short s[8]; } uz, un, uh;
  uz.q = *(const ux4*)(zp + i8);
  un.q = *(const ux4*)(np + i8);
#pragma unroll
  for (int i = 0; i < 8; ++i) {
    float z = fast_sigmoid(bfbits2f(uz.s[i]) + bz[c0 + i]);
    float n = fast_tanh(bfbits2f(un.s[i]) + bn[c0 + i]);
    uh.s[i] = f2bfbits((1.f - z) * n);
  }
  *(ux4*)(h + i8) = uh.q;
}

// ---------------------------------------------------------------------------
extern "C" void kernel_launch(void* const* d_in, const int* in_sizes, int n_in,
                              void* d_out, int out_size, void* d_ws,
                              size_t ws_size, hipStream_t stream) {
  (void)in_sizes; (void)n_in; (void)out_size; (void)ws_size;
  const float* x      = (const float*)d_in[0];   // [16][128][1024]
  const float* w_px   = (const float*)d_in[1];   // [256][128]
  const float* w_conv = (const float*)d_in[2];   // [256][512][3][3]
  const float* wq     = (const float*)d_in[3];   // [256][256]
  const float* wk     = (const float*)d_in[4];
  const float* wv     = (const float*)d_in[5];
  const float* s      = (const float*)d_in[6];   // [256]
  const float* wz     = (const float*)d_in[7];   // [256][512]
  const float* bzb    = (const float*)d_in[8];
  const float* wn     = (const float*)d_in[9];
  const float* bnb    = (const float*)d_in[10];
  const float* wout   = (const float*)d_in[11];  // [128][256]
  const float* bout   = (const float*)d_in[12];

  size_t off = 0;
  auto take = [&](size_t bytes) {
    void* p = (char*)d_ws + off;
    off += (bytes + 255) & ~(size_t)255;
    return p;
  };
  __bf16* wpx_p = (__bf16*)take(256 * 128 * 2);
  __bf16* wcv_p = (__bf16*)take(9 * 256 * 256 * 2);
  __bf16* wq_p  = (__bf16*)take(256 * 256 * 2);
  __bf16* wk_p  = (__bf16*)take(256 * 256 * 2);
  __bf16* wv_p  = (__bf16*)take(256 * 256 * 2);
  __bf16* wz_p  = (__bf16*)take(256 * 512 * 2);
  __bf16* wn_p  = (__bf16*)take(256 * 512 * 2);
  __bf16* wo_p  = (__bf16*)take(128 * 256 * 2);
  __bf16* xt = (__bf16*)take((size_t)16 * 1024 * 256 * 2);   // pixel-major
  __bf16* ca = (__bf16*)take((size_t)16 * 1024 * 512 * 2);   // [c | a]
  __bf16* qb = (__bf16*)take((size_t)16 * 1024 * 256 * 2);
  __bf16* kb = (__bf16*)take((size_t)16 * 1024 * 256 * 2);
  __bf16* vb = (__bf16*)take((size_t)16 * 1024 * 256 * 2);
  float*  Sc = (float*)take((size_t)16 * 1024 * 1024 * 4);   // scores f32
  __bf16* Pb = (__bf16*)take((size_t)16 * 1024 * 1024 * 2);  // probs bf16
  // zpre/npre/h alias the score buffer (dead after softmax):
  __bf16* zpre = (__bf16*)Sc;
  __bf16* npre = (__bf16*)Sc + (size_t)16 * 1024 * 256;
  __bf16* hb   = (__bf16*)Sc + (size_t)32 * 1024 * 256;

  // ---- weight repack f32 -> bf16 ----
  pack_bf16<<<(32768 + 255) / 256, 256, 0, stream>>>(w_px, wpx_p, 32768);
  pack_conv9<<<(589824 + 255) / 256, 256, 0, stream>>>(w_conv, wcv_p);
  pack_bf16<<<(65536 + 255) / 256, 256, 0, stream>>>(wq, wq_p, 65536);
  pack_bf16<<<(65536 + 255) / 256, 256, 0, stream>>>(wk, wk_p, 65536);
  pack_bf16<<<(65536 + 255) / 256, 256, 0, stream>>>(wv, wv_p, 65536);
  pack_bf16<<<(131072 + 255) / 256, 256, 0, stream>>>(wz, wz_p, 131072);
  pack_bf16<<<(131072 + 255) / 256, 256, 0, stream>>>(wn, wn_p, 131072);
  pack_bf16<<<(32768 + 255) / 256, 256, 0, stream>>>(wout, wo_p, 32768);

  // 1) xt = tanh(x @ w_px^T) : A fp32 NCHW gather, M=16384 N=256 K=128
  gemm_wmma<1, 0, 1><<<dim3(2, 128, 1), 256, 0, stream>>>(
      x, wpx_p, xt, 16384, 256, 128, /*lda=HW*/1024, 128, 256, 0, 0, 0, nullptr);

  // 2) c = conv3x3(xt) -> ca[:, :256] : implicit GEMM, K = 9*256
  gemm_wmma<2, 0, 0><<<dim3(2, 128, 1), 256, 0, stream>>>(
      xt, wcv_p, ca, 16384, 256, 2304, 256, 2304, 512, 0, 0, 0, nullptr);

  // 3) q,k,v = c @ w^T (first 256 cols of ca, lda=512)
  gemm_wmma<0, 0, 0><<<dim3(2, 128, 1), 256, 0, stream>>>(
      ca, wq_p, qb, 16384, 256, 256, 512, 256, 256, 0, 0, 0, nullptr);
  gemm_wmma<0, 0, 0><<<dim3(2, 128, 1), 256, 0, stream>>>(
      ca, wk_p, kb, 16384, 256, 256, 512, 256, 256, 0, 0, 0, nullptr);
  gemm_wmma<0, 0, 0><<<dim3(2, 128, 1), 256, 0, stream>>>(
      ca, wv_p, vb, 16384, 256, 256, 512, 256, 256, 0, 0, 0, nullptr);

  // 4) L2-normalize q (with scale s) and k over channels
  l2norm_scale<<<2048, 256, 0, stream>>>(qb, s);
  l2norm_scale<<<2048, 256, 0, stream>>>(kb, nullptr);

  // 5) scores S[n,qp,dp] = q . k (batched 16; M=N=1024, K=256) -> f32
  gemm_wmma<0, 0, 2><<<dim3(8, 8, 16), 256, 0, stream>>>(
      qb, kb, Sc, 1024, 1024, 256, 256, 256, 1024,
      262144LL, 262144LL, 1048576LL, nullptr);

  // 6) row softmax -> bf16 probabilities
  softmax_row<<<16384, 256, 0, stream>>>(Sc, Pb);

  // 7) a = P @ v -> ca[:, 256:512] (B = v pixel-major, transposed staging)
  gemm_wmma<0, 1, 0><<<dim3(2, 8, 16), 256, 0, stream>>>(
      Pb, vb, ca + 256, 1024, 256, 1024, 1024, /*N-stride*/256, 512,
      1048576LL, 262144LL, 524288LL, nullptr);

  // 8) z/n pre-activations over ca (K=512)
  gemm_wmma<0, 0, 0><<<dim3(2, 128, 1), 256, 0, stream>>>(
      ca, wz_p, zpre, 16384, 256, 512, 512, 512, 256, 0, 0, 0, nullptr);
  gemm_wmma<0, 0, 0><<<dim3(2, 128, 1), 256, 0, stream>>>(
      ca, wn_p, npre, 16384, 256, 512, 512, 512, 256, 0, 0, 0, nullptr);

  // 9) h = (1 - sigmoid(zpre+bz)) * tanh(npre+bn)
  gru_combine<<<2048, 256, 0, stream>>>(zpre, npre, bzb, bnb, hb);

  // 10) out = h @ wout^T + bout, stored f32 NCHW [16][128][1024]
  gemm_wmma<0, 0, 3><<<dim3(1, 128, 1), 256, 0, stream>>>(
      hb, wo_p, d_out, 16384, 128, 256, 256, 256, /*HW*/1024, 0, 0, 0, bout);
}
